// MultiheadRetention_27797028340014
// MI455X (gfx1250) — matrix operand order repaired and checked by reading, not compiled, not run on hardware
//
#include <hip/hip_runtime.h>
#include <hip/hip_bf16.h>

// ---------------------------------------------------------------------------
// Multihead Retention forward for gfx1250 (MI455X).
// bf16 WMMA + f32 accumulate, TDM (tensor_load_to_lds) double-buffered staging.
// B=1, N=4096, E=512, H=8, HD=64.
// ---------------------------------------------------------------------------

#define NSEQ 4096
#define EDIM 512
#define HEADS 8
#define HD 64

typedef __attribute__((ext_vector_type(16))) __bf16 v16bf;
typedef __attribute__((ext_vector_type(8)))  float  v8f;

#define WMMA_BF16(a, b, c) \
  __builtin_amdgcn_wmma_f32_16x16x32_bf16(false, (a), false, (b), (short)0, (c), false, false)

// --- Tensor Data Mover: 2D tile (rows x cols bf16) from a row-major matrix
//     with row stride EDIM into contiguous LDS. Descriptor per CDNA5 ISA ch.8.
static __device__ __forceinline__ unsigned lds_addr32(const void* p) {
  // flat LDS addresses map to LDS offset via addr[31:0] (ISA 10.2)
  return (unsigned)(uintptr_t)p;
}

static __device__ __forceinline__ void tdm_load_2d(const __bf16* gsrc, unsigned lds_off,
                                                   unsigned tile_cols, unsigned tile_rows) {
  typedef __attribute__((ext_vector_type(4))) unsigned tv4u;
  typedef __attribute__((ext_vector_type(8))) int      tv8i;
  typedef __attribute__((ext_vector_type(4))) int      tv4i;
  const unsigned long long ga = (unsigned long long)(uintptr_t)gsrc;
  tv4u g0;
  g0[0] = 1u;                                           // count=1 (valid), user mode
  g0[1] = lds_off;                                      // lds_addr (bytes)
  g0[2] = (unsigned)(ga & 0xFFFFFFFFull);               // global_addr[31:0]
  g0[3] = (unsigned)((ga >> 32) & 0x01FFFFFFull)        // global_addr[56:32]
          | (2u << 30);                                 // type = 2 ("image")
  tv8i g1;
  g1[0] = (int)(1u << 16);                              // data_size=1 -> 2 bytes; mask=0
  g1[1] = (int)(((unsigned)EDIM & 0xFFFFu) << 16);      // tensor_dim0[15:0] @ [63:48]
  g1[2] = (int)((((unsigned)EDIM >> 16) & 0xFFFFu)      // tensor_dim0[31:16]
          | (((unsigned)NSEQ & 0xFFFFu) << 16));        // tensor_dim1[15:0]
  g1[3] = (int)((((unsigned)NSEQ >> 16) & 0xFFFFu)      // tensor_dim1[31:16]
          | (tile_cols << 16));                         // tile_dim0
  g1[4] = (int)tile_rows;                               // tile_dim1 (tile_dim2=0)
  g1[5] = (int)EDIM;                                    // tensor_dim0_stride[31:0]
  g1[6] = 0;                                            // stride hi / dim1_stride lo
  g1[7] = 0;
  tv4i gz = {0, 0, 0, 0};
#if __clang_major__ >= 23
  tv8i gz8 = {0, 0, 0, 0, 0, 0, 0, 0};
  __builtin_amdgcn_tensor_load_to_lds(g0, g1, gz, gz, gz8, 0);
#else
  __builtin_amdgcn_tensor_load_to_lds(g0, g1, gz, gz, 0);
#endif
}

// --- fragment gathers (contiguous k order: lane half h holds k = 16h..16h+15, wave32) ----------------
// A (16x32 bf16): lane m = lane&15; lanes>=16 shift K by +16 (contiguous 16).
static __device__ __forceinline__ v16bf frag_a(const __bf16* lds, int row0, int ld, int k0) {
  const int lane = threadIdx.x & 31;
  const int m    = lane & 15;
  const int koff = (lane >> 4) << 4;
  const __bf16* p = lds + (size_t)(row0 + m) * ld + k0 + koff;
  v16bf a;
#pragma unroll
  for (int v = 0; v < 16; ++v) a[v] = p[v];
  return a;
}

// B (32x16 bf16), LDS tile stored T[n][k]: B[k,n] = T[n0+n][k0+k]. Contiguous 32B.
static __device__ __forceinline__ v16bf frag_b_nk(const __bf16* lds, int n0, int ld, int k0) {
  const int lane = threadIdx.x & 31;
  const int n  = lane & 15;
  const int kh = (lane >> 4) << 4;
  const __bf16* p = lds + (size_t)(n0 + n) * ld + k0 + kh;
  v16bf b;
#pragma unroll
  for (int t = 0; t < 16; ++t) b[t] = p[t];
  return b;
}

// B (32x16 bf16), LDS tile stored T[k][n]: B[k,n] = T[k0+k][n0+n].
static __device__ __forceinline__ v16bf frag_b_kn(const __bf16* lds, int k0, int ld, int n0) {
  const int lane = threadIdx.x & 31;
  const int n  = lane & 15;
  const int kh = (lane >> 4) << 4;
  v16bf b;
#pragma unroll
  for (int t = 0; t < 16; ++t) b[t] = lds[(size_t)(k0 + kh + t) * ld + n0 + n];
  return b;
}

// --- fp32 -> bf16 conversion ------------------------------------------------
__global__ __launch_bounds__(256) void cvt_bf16(const float* __restrict__ src,
                                                __bf16* __restrict__ dst, int n) {
  int i = blockIdx.x * 256 + threadIdx.x;
  if (i < n) dst[i] = (__bf16)src[i];
}

// --- projection GEMM: C[m,n] = sum_k X[m,k]*W[n,k] + bias[n] ---------------
// Block = 256 threads (8 waves). Tile: 128(M) x 64(N), K stepped by 32,
// TDM double-buffered staging of A and W tiles.
// emode: 0 -> bf16 out, 2 -> fp32 silu(out) (gate), 3 -> fp32 out.
__global__ __launch_bounds__(256) void proj_gemm(const __bf16* __restrict__ X,
                                                 const __bf16* __restrict__ W,
                                                 const float* __restrict__ bias,
                                                 void* __restrict__ out, int emode) {
  __shared__ __bf16 ldsA[2][128 * 32];
  __shared__ __bf16 ldsW[2][64 * 32];
  const int t = threadIdx.x;
  const int wave = t >> 5, lane = t & 31;
  const int m0 = blockIdx.x * 128;
  const int n0 = blockIdx.y * 64;
  v8f acc[4] = {};

  if (t < 32) {  // prologue: stage first tiles via TDM (issued once by wave 0)
    tdm_load_2d(X + (size_t)m0 * EDIM, lds_addr32(ldsA[0]), 32, 128);
    tdm_load_2d(W + (size_t)n0 * EDIM, lds_addr32(ldsW[0]), 32, 64);
    __builtin_amdgcn_s_wait_tensorcnt(0);
  }
  __syncthreads();

  int buf = 0;
  for (int kk = 0; kk < EDIM; kk += 32) {
    if (t < 32 && kk + 32 < EDIM) {  // prefetch next K tiles into the other buffer
      tdm_load_2d(X + (size_t)m0 * EDIM + kk + 32, lds_addr32(ldsA[buf ^ 1]), 32, 128);
      tdm_load_2d(W + (size_t)n0 * EDIM + kk + 32, lds_addr32(ldsW[buf ^ 1]), 32, 64);
    }
    v16bf a = frag_a(ldsA[buf], wave * 16, 32, 0);
#pragma unroll
    for (int nt = 0; nt < 4; ++nt) {
      v16bf b = frag_b_nk(ldsW[buf], nt * 16, 32, 0);
      acc[nt] = WMMA_BF16(a, b, acc[nt]);
    }
    if (t < 32) __builtin_amdgcn_s_wait_tensorcnt(0);
    __syncthreads();
    buf ^= 1;
  }

  const int hi8 = (lane >> 4) << 3;
  const int nl  = lane & 15;
#pragma unroll
  for (int nt = 0; nt < 4; ++nt) {
#pragma unroll
    for (int r = 0; r < 8; ++r) {
      const int grow = m0 + wave * 16 + r + hi8;
      const int gcol = n0 + nt * 16 + nl;
      float v = acc[nt][r] + bias[gcol];
      const size_t idx = (size_t)grow * EDIM + gcol;
      if (emode == 0) {
        ((__bf16*)out)[idx] = (__bf16)v;
      } else if (emode == 2) {
        ((float*)out)[idx] = v / (1.0f + __expf(-v));  // silu
      } else {
        ((float*)out)[idx] = v;
      }
    }
  }
}

// --- retention core + GroupNorm + gate -------------------------------------
// grid = (N/64, H); block = 128 threads (4 waves), each wave owns a 16-row strip.
// K/V blocks staged by TDM with double buffering; windowed decay scan.
__global__ __launch_bounds__(128) void retention_kernel(const __bf16* __restrict__ Qb,
                                                        const __bf16* __restrict__ Kb,
                                                        const __bf16* __restrict__ Vb,
                                                        const float* __restrict__ Gate,
                                                        __bf16* __restrict__ RG) {
  __shared__ __bf16 qs[64 * 64];
  __shared__ __bf16 ks[2][64 * 64];
  __shared__ __bf16 vs[2][64 * 64];
  __shared__ __bf16 ss[64 * 64];
  __shared__ float  rets[64 * 64];

  const int t = threadIdx.x, wave = t >> 5, lane = t & 31;
  const int i0 = blockIdx.x * 64;
  const int h  = blockIdx.y;
  const int hi8 = (lane >> 4) << 3;
  const int nl  = lane & 15;

  // gamma_h = 1 - exp(linspace(log(1/32), log(1/512), 8))[h]
  const float x0 = logf(1.0f / 32.0f), x1 = logf(1.0f / 512.0f);
  const float gamma = 1.0f - __expf(x0 + (x1 - x0) * ((float)h / 7.0f));
  const float log2g = log2f(gamma);

  // window: skip key blocks whose contribution gamma^dist < 2^-44
  int cap = (int)(-44.0f / log2g) + 64;
  int jstart = i0 - cap;
  if (jstart < 0) jstart = 0;
  jstart &= ~63;

  if (t < 32) {  // stage Q strip + first K/V blocks via TDM
    tdm_load_2d(Qb + (size_t)i0 * EDIM + h * HD, lds_addr32(qs), 64, 64);
    tdm_load_2d(Kb + (size_t)jstart * EDIM + h * HD, lds_addr32(ks[0]), 64, 64);
    tdm_load_2d(Vb + (size_t)jstart * EDIM + h * HD, lds_addr32(vs[0]), 64, 64);
    __builtin_amdgcn_s_wait_tensorcnt(0);
  }
  __syncthreads();

  const v16bf aq0 = frag_a(qs, wave * 16, 64, 0);
  const v16bf aq1 = frag_a(qs, wave * 16, 64, 32);
  v8f acc[4] = {};

  int buf = 0;
  for (int j0 = jstart; j0 <= i0; j0 += 64) {
    if (t < 32 && j0 + 64 <= i0) {  // prefetch next K/V blocks
      tdm_load_2d(Kb + (size_t)(j0 + 64) * EDIM + h * HD, lds_addr32(ks[buf ^ 1]), 64, 64);
      tdm_load_2d(Vb + (size_t)(j0 + 64) * EDIM + h * HD, lds_addr32(vs[buf ^ 1]), 64, 64);
    }
    const __bf16* kc = ks[buf];
    const __bf16* vc = vs[buf];

    // S = Q @ K^T over d (HD=64 -> two k=32 WMMA steps), then decay mask.
#pragma unroll
    for (int js = 0; js < 4; ++js) {
      v8f s = {};
      v16bf b0 = frag_b_nk(kc, js * 16, 64, 0);   // B[d,j] = K[j][d]
      v16bf b1 = frag_b_nk(kc, js * 16, 64, 32);
      s = WMMA_BF16(aq0, b0, s);
      s = WMMA_BF16(aq1, b1, s);
#pragma unroll
      for (int r = 0; r < 8; ++r) {
        const int li = wave * 16 + r + hi8;
        const int lj = js * 16 + nl;
        const float dist = (float)((i0 + li) - (j0 + lj));
        const float w = (dist >= 0.0f) ? exp2f(dist * log2g) : 0.0f;
        ss[li * 64 + lj] = (__bf16)(s[r] * w);
      }
    }
    __syncthreads();

    // ret += S @ V over j (64 -> two k=32 WMMA steps)
    const v16bf as0 = frag_a(ss, wave * 16, 64, 0);
    const v16bf as1 = frag_a(ss, wave * 16, 64, 32);
#pragma unroll
    for (int dsub = 0; dsub < 4; ++dsub) {
      v16bf bv0 = frag_b_kn(vc, 0, 64, dsub * 16);   // B[j,d] = V[j][d]
      v16bf bv1 = frag_b_kn(vc, 32, 64, dsub * 16);
      acc[dsub] = WMMA_BF16(as0, bv0, acc[dsub]);
      acc[dsub] = WMMA_BF16(as1, bv1, acc[dsub]);
    }

    if (t < 32) __builtin_amdgcn_s_wait_tensorcnt(0);
    __syncthreads();
    buf ^= 1;
  }

  // dump accumulators for the GroupNorm epilogue
#pragma unroll
  for (int dsub = 0; dsub < 4; ++dsub)
#pragma unroll
    for (int r = 0; r < 8; ++r)
      rets[(wave * 16 + r + hi8) * 64 + dsub * 16 + nl] = acc[dsub][r];
  __syncthreads();

  // GroupNorm over HD (biased var) + SiLU gate, write bf16 for final GEMM
  if (t < 64) {
    const int row = t;
    float mu = 0.0f;
    for (int d = 0; d < HD; ++d) mu += rets[row * 64 + d];
    mu *= (1.0f / 64.0f);
    float var = 0.0f;
    for (int d = 0; d < HD; ++d) {
      const float dv = rets[row * 64 + d] - mu;
      var += dv * dv;
    }
    var *= (1.0f / 64.0f);
    const float inv = rsqrtf(var + 1e-6f);
    for (int d = 0; d < HD; ++d) {
      const float v = (rets[row * 64 + d] - mu) * inv;
      const float g = Gate[(size_t)(i0 + row) * EDIM + h * HD + d];
      RG[(size_t)(i0 + row) * EDIM + h * HD + d] = (__bf16)(v * g);
    }
  }
}

// ---------------------------------------------------------------------------
extern "C" void kernel_launch(void* const* d_in, const int* in_sizes, int n_in,
                              void* d_out, int out_size, void* d_ws, size_t ws_size,
                              hipStream_t stream) {
  (void)in_sizes; (void)n_in; (void)out_size; (void)ws_size;
  const float* query = (const float*)d_in[0];
  const float* key   = (const float*)d_in[1];
  const float* value = (const float*)d_in[2];
  const float* Wq = (const float*)d_in[3];  const float* bq = (const float*)d_in[4];
  const float* Wk = (const float*)d_in[5];  const float* bk = (const float*)d_in[6];
  const float* Wv = (const float*)d_in[7];  const float* bv = (const float*)d_in[8];
  const float* Wg = (const float*)d_in[9];  const float* bg = (const float*)d_in[10];
  const float* Wo = (const float*)d_in[11]; const float* bo = (const float*)d_in[12];

  const int NE = NSEQ * EDIM;          // 2,097,152 elements
  const int WE = EDIM * EDIM;          // 262,144 elements
  char* ws = (char*)d_ws;
  size_t o = 0;
  auto carve = [&](size_t bytes) {
    char* p = ws + o;
    o += (bytes + 255) & ~(size_t)255;
    return p;
  };
  __bf16* xq = (__bf16*)carve((size_t)NE * 2);
  __bf16* xk = (__bf16*)carve((size_t)NE * 2);
  __bf16* xv = (__bf16*)carve((size_t)NE * 2);
  __bf16* wq = (__bf16*)carve((size_t)WE * 2);
  __bf16* wk = (__bf16*)carve((size_t)WE * 2);
  __bf16* wv = (__bf16*)carve((size_t)WE * 2);
  __bf16* wg = (__bf16*)carve((size_t)WE * 2);
  __bf16* wo = (__bf16*)carve((size_t)WE * 2);
  __bf16* Qb = (__bf16*)carve((size_t)NE * 2);
  __bf16* Kb = (__bf16*)carve((size_t)NE * 2);
  __bf16* Vb = (__bf16*)carve((size_t)NE * 2);
  float*  Gt = (float*)carve((size_t)NE * 4);
  __bf16* RG = (__bf16*)carve((size_t)NE * 2);

  // fp32 -> bf16 staging
  cvt_bf16<<<(NE + 255) / 256, 256, 0, stream>>>(query, xq, NE);
  cvt_bf16<<<(NE + 255) / 256, 256, 0, stream>>>(key,   xk, NE);
  cvt_bf16<<<(NE + 255) / 256, 256, 0, stream>>>(value, xv, NE);
  cvt_bf16<<<(WE + 255) / 256, 256, 0, stream>>>(Wq, wq, WE);
  cvt_bf16<<<(WE + 255) / 256, 256, 0, stream>>>(Wk, wk, WE);
  cvt_bf16<<<(WE + 255) / 256, 256, 0, stream>>>(Wv, wv, WE);
  cvt_bf16<<<(WE + 255) / 256, 256, 0, stream>>>(Wg, wg, WE);
  cvt_bf16<<<(WE + 255) / 256, 256, 0, stream>>>(Wo, wo, WE);

  // projections
  dim3 pg(NSEQ / 128, EDIM / 64);
  proj_gemm<<<pg, 256, 0, stream>>>(xq, wq, bq, Qb, 0);
  proj_gemm<<<pg, 256, 0, stream>>>(xk, wk, bk, Kb, 0);
  proj_gemm<<<pg, 256, 0, stream>>>(xv, wv, bv, Vb, 0);
  proj_gemm<<<pg, 256, 0, stream>>>(xq, wg, bg, Gt, 2);  // silu gate, fp32

  // retention + GroupNorm + gate
  retention_kernel<<<dim3(NSEQ / 64, HEADS), 128, 0, stream>>>(Qb, Kb, Vb, Gt, RG);

  // output projection (fp32 + bias) into d_out
  proj_gemm<<<pg, 256, 0, stream>>>(RG, wo, bo, d_out, 3);
}
